// GRUDecoder_17506286699167
// MI455X (gfx1250) — matrix operand order repaired and checked by reading, not compiled, not run on hardware
//
#include <hip/hip_runtime.h>
#include <hip/hip_bf16.h>

typedef __attribute__((ext_vector_type(16))) _Float16 v16h;
typedef __attribute__((ext_vector_type(8)))  _Float16 v8h;
typedef __attribute__((ext_vector_type(8)))  float    v8f;

#define N_NODES   200000
#define N_EDGES   1600000
#define N_GRAPHS  4096
#define NB        256
#define NT        16
#define HDIM      128

__device__ __forceinline__ float fast_sigmoid(float x) {
    return __builtin_amdgcn_rcpf(1.0f + __expf(-x));
}
__device__ __forceinline__ float fast_tanh(float x) {
#if __has_builtin(__builtin_amdgcn_tanhf)
    return __builtin_amdgcn_tanhf(x);
#else
    float e2 = __expf(2.0f * x);
    return 1.0f - 2.0f * __builtin_amdgcn_rcpf(e2 + 1.0f);
#endif
}

// ---------------------------------------------------------------------------
// Utility kernels
// ---------------------------------------------------------------------------
__global__ void k_zero_f32(float* __restrict__ p, int n) {
    int i = blockIdx.x * blockDim.x + threadIdx.x;
    if (i < n) p[i] = 0.0f;
}

__global__ void k_f32_to_f16(const float* __restrict__ s, _Float16* __restrict__ d, int n) {
    int i = blockIdx.x * blockDim.x + threadIdx.x;
    if (i < n) d[i] = (_Float16)s[i];
}

// ---------------------------------------------------------------------------
// Layer-0 conv GEMM (K=5): pure VALU, weights in LDS, f16 output
// ---------------------------------------------------------------------------
__global__ void k_gemm_k5(const float* __restrict__ A,   // [M,5] f32
                          const float* __restrict__ W,   // [5,64]
                          const float* __restrict__ bias,
                          const float* __restrict__ agg, // f32 or null
                          int do_relu, _Float16* __restrict__ out, int M) {
    __shared__ float wS[5 * 64];
    for (int i = threadIdx.x; i < 5 * 64; i += blockDim.x) wS[i] = W[i];
    __syncthreads();
    int idx = blockIdx.x * blockDim.x + threadIdx.x;
    if (idx >= M * 64) return;
    int m = idx >> 6, n = idx & 63;
    const float* ar = A + (size_t)m * 5;
    float v = 0.0f;
#pragma unroll
    for (int k = 0; k < 5; ++k) v += ar[k] * wS[k * 64 + n];
    if (agg)  v += agg[idx];
    if (bias) v += bias[n];
    if (do_relu) v = v > 0.0f ? v : 0.0f;
    out[idx] = (_Float16)v;
}

// ---------------------------------------------------------------------------
// WMMA GEMM: out[M,N] = A[M,K](f16) x W(f16) (+agg +bias, relu)
//   TRANSB=0: W stored [K,N];  TRANSB=1: W stored [N,K]
// Block = 16 rows. A tile (16 x K f16 = contiguous 2K..4KB) staged into LDS
// with gfx1250 async-copy (global_load_async_to_lds_b128, ASYNCcnt), then 8
// waves cover the column tiles; K loop fully unrolled.
// ---------------------------------------------------------------------------
template <typename OT, int K, int TRANSB>
__global__ void k_gemm_wmma(const _Float16* __restrict__ A,
                            const _Float16* __restrict__ W,
                            const float* __restrict__ bias,
                            const float* __restrict__ agg,
                            int do_relu, OT* __restrict__ out, int N) {
    __shared__ _Float16 aLds[16 * K];
    const int rowBase = blockIdx.x * 16;
    const int tid = threadIdx.x;

    // Async byte-copy of the contiguous 16xK f16 tile into LDS (16B chunks).
    constexpr int CHUNKS = (16 * K * 2) / 16;
    if (tid < CHUNKS) {
        const char* gsrc = (const char*)(A + (size_t)rowBase * K) + tid * 16;
        unsigned loff = (unsigned)(size_t)(void*)aLds + tid * 16;
        asm volatile("global_load_async_to_lds_b128 %0, %1, off"
                     :: "v"(loff), "v"(gsrc) : "memory");
    }
    asm volatile("s_wait_asynccnt 0x0" ::: "memory");
    __syncthreads();

    const int wave = tid >> 5;
    const int lane = tid & 31;
    const int mloc = lane & 15;          // A row / D column position
    const int hi   = lane >> 4;          // lane-half select
    const int aBase = hi * 8;            // unused; A/B fragments use contiguous k = kk + 16*hi + j
    const int nTiles = N >> 4;

    for (int ct = wave; ct < nTiles; ct += 8) {
        const int nGlob = ct * 16 + mloc;
        v8f c = {};
#pragma unroll
        for (int kk = 0; kk < K; kk += 32) {
            v16h a, b;
#pragma unroll
            for (int j = 0; j < 16; ++j) {
                int ka = kk + hi * 16 + j;
                a[j] = aLds[mloc * K + ka];
                int kb = kk + hi * 16 + j;                 // B: col per lane pair
                b[j] = TRANSB ? W[(size_t)nGlob * K + kb]
                              : W[(size_t)kb * N + nGlob];
            }
            c = __builtin_amdgcn_wmma_f32_16x16x32_f16(false, a, false, b,
                                                       (short)0, c, false, false);
        }
#pragma unroll
        for (int r = 0; r < 8; ++r) {
            int mGlob = rowBase + r + hi * 8;              // D row layout
            size_t o = (size_t)mGlob * N + nGlob;
            float v = c[r];
            if (agg)  v += agg[o];
            if (bias) v += bias[nGlob];
            if (do_relu) v = v > 0.0f ? v : 0.0f;
            out[o] = (OT)v;
        }
    }
}

// ---------------------------------------------------------------------------
// Edge scatter: agg[dst] += edge_w * hn[src].  hn is f16 (51MB: L2-resident),
// agg is f32 with global_atomic_add_f32 (resolves in the 192MB L2).
// One thread = one (edge, 32-feature chunk)
// ---------------------------------------------------------------------------
__global__ void k_edge_scatter(const _Float16* __restrict__ hn,
                               const float* __restrict__ ea,
                               const int* __restrict__ ei,
                               int C, float* __restrict__ agg) {
    const int chunks = C >> 5;
    int idx = blockIdx.x * blockDim.x + threadIdx.x;
    if (idx >= N_EDGES * chunks) return;
    int e = idx / chunks;
    int q = (idx - e * chunks) << 5;
    int s = ei[e], d = ei[N_EDGES + e];
    float w = ea[e];
    const v8h* src = (const v8h*)(hn + (size_t)s * C + q);
    float* dst = agg + (size_t)d * C + q;
#pragma unroll
    for (int i = 0; i < 4; ++i) {
        v8h v = src[i];
#pragma unroll
        for (int j = 0; j < 8; ++j)
            atomicAdd(dst + i * 8 + j, (float)v[j] * w);
    }
}

// ---------------------------------------------------------------------------
// Mean-pool accumulate + finalize/gather (h3 is f16, sums f32, seq f16)
// ---------------------------------------------------------------------------
__global__ void k_pool_accum(const _Float16* __restrict__ h,
                             const int* __restrict__ labels,
                             float* __restrict__ sums, float* __restrict__ cnt) {
    int idx = blockIdx.x * blockDim.x + threadIdx.x;      // node * 16 chunks
    if (idx >= N_NODES * 16) return;
    int node = idx >> 4;
    int f = (idx & 15) << 3;
    int g = labels[node];
    v8h v = *(const v8h*)(h + (size_t)node * HDIM + f);
    float* dst = sums + (size_t)g * HDIM + f;
#pragma unroll
    for (int j = 0; j < 8; ++j) atomicAdd(dst + j, (float)v[j]);
    if (f == 0) atomicAdd(&cnt[g], 1.0f);
}

__global__ void k_pool_gather(const float* __restrict__ sums,
                              const float* __restrict__ cnt,
                              const int* __restrict__ lmap,
                              _Float16* __restrict__ seq) {
    int idx = blockIdx.x * blockDim.x + threadIdx.x;      // 4096*128
    if (idx >= N_GRAPHS * HDIM) return;
    int i = idx >> 7, f = idx & 127;
    int g = lmap[i];
    float c = cnt[g]; if (c < 1.0f) c = 1.0f;
    seq[idx] = (_Float16)(sums[(size_t)g * HDIM + f] / c);
}

// ---------------------------------------------------------------------------
// GRU recurrence: one block owns 16 batch rows; h in LDS (f32 + f16 mirror).
// Per step: gh = h @ whh^T via WMMA (24 tiles over 8 waves), then gates with
// v_rcp/v_tanh fast paths. Writes both f32 (decoder) and f16 (next GEMM A).
// ---------------------------------------------------------------------------
__global__ void k_gru_seq(const float* __restrict__ xproj,    // [B*T, 384]
                          const _Float16* __restrict__ whh16, // [384, 128]
                          const float* __restrict__ bhh,      // [384]
                          float* __restrict__ outF32,         // [B*T, 128]
                          _Float16* __restrict__ outF16) {    // [B*T, 128]
    __shared__ float    hF32[16 * HDIM];
    __shared__ _Float16 hF16[16 * HDIM];
    __shared__ float    gh[16 * 384];
    const int tid = threadIdx.x;
    const int bBase = blockIdx.x * 16;
    for (int i = tid; i < 16 * HDIM; i += 256) {
        hF32[i] = 0.0f; hF16[i] = (_Float16)0.0f;
    }
    __syncthreads();

    const int wave = tid >> 5, lane = tid & 31;
    const int mloc = lane & 15, hi = lane >> 4, aBase = hi * 8;

    for (int t = 0; t < NT; ++t) {
        // gh = h @ whh^T  (16 x 384)
        for (int ct = wave; ct < 24; ct += 8) {
            const int nGlob = ct * 16 + mloc;
            v8f c = {};
#pragma unroll
            for (int kk = 0; kk < HDIM; kk += 32) {
                v16h a, b;
#pragma unroll
                for (int j = 0; j < 16; ++j) {
                    int ka = kk + hi * 16 + j;
                    a[j] = hF16[mloc * HDIM + ka];
                    int kb = kk + hi * 16 + j;
                    b[j] = whh16[(size_t)nGlob * HDIM + kb];  // B[k][n]=whh[n][k]
                }
                c = __builtin_amdgcn_wmma_f32_16x16x32_f16(false, a, false, b,
                                                           (short)0, c, false, false);
            }
#pragma unroll
            for (int r = 0; r < 8; ++r)
                gh[(r + hi * 8) * 384 + nGlob] = c[r];
        }
        __syncthreads();

        // gates: 16 rows x 128 units = 2048 elems / 256 threads
#pragma unroll
        for (int u = 0; u < 8; ++u) {
            int elem = tid + u * 256;
            int row = elem >> 7, j = elem & 127;
            int m = (bBase + row) * NT + t;
            const float* xp = xproj + (size_t)m * 384;
            float xr = xp[j], xz = xp[128 + j], xn = xp[256 + j];
            float hr = gh[row * 384 + j]        + bhh[j];
            float hz = gh[row * 384 + 128 + j]  + bhh[128 + j];
            float hn = gh[row * 384 + 256 + j]  + bhh[256 + j];
            float r_ = fast_sigmoid(xr + hr);
            float z_ = fast_sigmoid(xz + hz);
            float n_ = fast_tanh(xn + r_ * hn);
            float hp = hF32[row * HDIM + j];
            float hv = (1.0f - z_) * n_ + z_ * hp;
            hF32[row * HDIM + j] = hv;
            hF16[row * HDIM + j] = (_Float16)hv;
            outF32[(size_t)m * HDIM + j] = hv;
            outF16[(size_t)m * HDIM + j] = (_Float16)hv;
        }
        __syncthreads();
    }
}

// ---------------------------------------------------------------------------
// Decoder: sigmoid(o2 @ w_dec + b_dec). One wave per output, shuffle reduce.
// outputs [0,4096) = predictions [B,T]; [4096,4352) = final (h_last = t=15)
// ---------------------------------------------------------------------------
__global__ void k_decoder(const float* __restrict__ o2,
                          const float* __restrict__ wdec,
                          const float* __restrict__ bdec,
                          float* __restrict__ out) {
    int o = blockIdx.x * (blockDim.x >> 5) + (threadIdx.x >> 5);
    int lane = threadIdx.x & 31;
    if (o >= N_GRAPHS + NB) return;
    const float* src = (o < N_GRAPHS)
        ? o2 + (size_t)o * HDIM
        : o2 + (size_t)((o - N_GRAPHS) * NT + (NT - 1)) * HDIM;
    float s = 0.0f;
#pragma unroll
    for (int i = 0; i < 4; ++i) {
        int k = lane * 4 + i;
        s += src[k] * wdec[k];
    }
    for (int off = 16; off > 0; off >>= 1) s += __shfl_down(s, off, 32);
    if (lane == 0) out[o] = fast_sigmoid(s + bdec[0]);
}

// ---------------------------------------------------------------------------
extern "C" void kernel_launch(void* const* d_in, const int* in_sizes, int n_in,
                              void* d_out, int out_size, void* d_ws, size_t ws_size,
                              hipStream_t stream) {
    (void)in_sizes; (void)n_in; (void)out_size; (void)ws_size;
    const float* x        = (const float*)d_in[0];
    const float* ea       = (const float*)d_in[1];
    const int*   ei       = (const int*)  d_in[2];
    const int*   labels   = (const int*)  d_in[3];
    const int*   lmap     = (const int*)  d_in[4];
    const float* w_self0  = (const float*)d_in[5];
    const float* w_nei0   = (const float*)d_in[6];
    const float* b0       = (const float*)d_in[7];
    const float* w_self1  = (const float*)d_in[8];
    const float* w_nei1   = (const float*)d_in[9];
    const float* b1       = (const float*)d_in[10];
    const float* w_self2  = (const float*)d_in[11];
    const float* w_nei2   = (const float*)d_in[12];
    const float* b2       = (const float*)d_in[13];
    const float* wih0     = (const float*)d_in[14];
    const float* whh0     = (const float*)d_in[15];
    const float* bih0     = (const float*)d_in[16];
    const float* bhh0     = (const float*)d_in[17];
    const float* wih1     = (const float*)d_in[18];
    const float* whh1     = (const float*)d_in[19];
    const float* bih1     = (const float*)d_in[20];
    const float* bhh1     = (const float*)d_in[21];
    const float* w_dec    = (const float*)d_in[22];
    const float* b_dec    = (const float*)d_in[23];
    float* out = (float*)d_out;

    // workspace layout
    char* ws = (char*)d_ws;
    size_t off = 0;
    auto alloc = [&](size_t bytes) -> void* {
        void* p = ws + off;
        off += (bytes + 255) & ~(size_t)255;
        return p;
    };
    _Float16* hA    = (_Float16*)alloc((size_t)N_NODES * 128 * 2);   // activations
    _Float16* hB    = (_Float16*)alloc((size_t)N_NODES * 128 * 2);
    float*    agg   = (float*)   alloc((size_t)N_NODES * 128 * 4);
    float*    sums  = (float*)   alloc((size_t)N_GRAPHS * HDIM * 4);
    float*    cnt   = (float*)   alloc((size_t)N_GRAPHS * 4);
    _Float16* seq   = (_Float16*)alloc((size_t)N_GRAPHS * HDIM * 2);
    float*    xproj = (float*)   alloc((size_t)N_GRAPHS * 384 * 4);
    float*    o1_32 = (float*)   alloc((size_t)N_GRAPHS * HDIM * 4);
    _Float16* o1_16 = (_Float16*)alloc((size_t)N_GRAPHS * HDIM * 2);
    float*    o2_32 = (float*)   alloc((size_t)N_GRAPHS * HDIM * 4);
    _Float16* o2_16 = (_Float16*)alloc((size_t)N_GRAPHS * HDIM * 2);
    _Float16* wn1_16  = (_Float16*)alloc(64 * 128 * 2);
    _Float16* ws1_16  = (_Float16*)alloc(64 * 128 * 2);
    _Float16* wn2_16  = (_Float16*)alloc(128 * 128 * 2);
    _Float16* ws2_16  = (_Float16*)alloc(128 * 128 * 2);
    _Float16* wih0_16 = (_Float16*)alloc(384 * 128 * 2);
    _Float16* whh0_16 = (_Float16*)alloc(384 * 128 * 2);
    _Float16* wih1_16 = (_Float16*)alloc(384 * 128 * 2);
    _Float16* whh1_16 = (_Float16*)alloc(384 * 128 * 2);

    auto cvt = [&](const float* s, _Float16* d, int n) {
        k_f32_to_f16<<<(n + 255) / 256, 256, 0, stream>>>(s, d, n);
    };
    cvt(w_nei1, wn1_16, 64 * 128);   cvt(w_self1, ws1_16, 64 * 128);
    cvt(w_nei2, wn2_16, 128 * 128);  cvt(w_self2, ws2_16, 128 * 128);
    cvt(wih0, wih0_16, 384 * 128);   cvt(whh0, whh0_16, 384 * 128);
    cvt(wih1, wih1_16, 384 * 128);   cvt(whh1, whh1_16, 384 * 128);

    const int rowBlocks = N_NODES / 16;   // 12500

    // ---- conv layer 0 (K=5, out=64): VALU GEMMs, f16 outputs ----
    k_zero_f32<<<(N_NODES * 64 + 255) / 256, 256, 0, stream>>>(agg, N_NODES * 64);
    k_gemm_k5<<<(N_NODES * 64 + 255) / 256, 256, 0, stream>>>(
        x, w_nei0, nullptr, nullptr, 0, hB, N_NODES);           // hn0 f16
    k_edge_scatter<<<(N_EDGES * 2 + 255) / 256, 256, 0, stream>>>(hB, ea, ei, 64, agg);
    k_gemm_k5<<<(N_NODES * 64 + 255) / 256, 256, 0, stream>>>(
        x, w_self0, b0, agg, 1, hA, N_NODES);                   // h1 [N,64] f16

    // ---- conv layer 1 (K=64, out=128): WMMA ----
    k_zero_f32<<<(N_NODES * 128 + 255) / 256, 256, 0, stream>>>(agg, N_NODES * 128);
    k_gemm_wmma<_Float16, 64, 0><<<rowBlocks, 256, 0, stream>>>(
        hA, wn1_16, nullptr, nullptr, 0, hB, 128);              // hn f16
    k_edge_scatter<<<(N_EDGES * 4 + 255) / 256, 256, 0, stream>>>(hB, ea, ei, 128, agg);
    k_gemm_wmma<_Float16, 64, 0><<<rowBlocks, 256, 0, stream>>>(
        hA, ws1_16, b1, agg, 1, hB, 128);                       // h2 [N,128] f16

    // ---- conv layer 2 (K=128, out=128): WMMA ----
    k_zero_f32<<<(N_NODES * 128 + 255) / 256, 256, 0, stream>>>(agg, N_NODES * 128);
    k_gemm_wmma<_Float16, 128, 0><<<rowBlocks, 256, 0, stream>>>(
        hB, wn2_16, nullptr, nullptr, 0, hA, 128);              // hn f16
    k_edge_scatter<<<(N_EDGES * 4 + 255) / 256, 256, 0, stream>>>(hA, ea, ei, 128, agg);
    k_gemm_wmma<_Float16, 128, 0><<<rowBlocks, 256, 0, stream>>>(
        hB, ws2_16, b2, agg, 1, hA, 128);                       // h3 [N,128] f16

    // ---- mean pool + gather into sequences ----
    k_zero_f32<<<(N_GRAPHS * HDIM + 255) / 256, 256, 0, stream>>>(sums, N_GRAPHS * HDIM);
    k_zero_f32<<<(N_GRAPHS + 255) / 256, 256, 0, stream>>>(cnt, N_GRAPHS);
    k_pool_accum<<<(N_NODES * 16 + 255) / 256, 256, 0, stream>>>(hA, labels, sums, cnt);
    k_pool_gather<<<(N_GRAPHS * HDIM + 255) / 256, 256, 0, stream>>>(sums, cnt, lmap, seq);

    // ---- GRU layer 0 ----
    k_gemm_wmma<float, 128, 1><<<N_GRAPHS / 16, 256, 0, stream>>>(
        seq, wih0_16, bih0, nullptr, 0, xproj, 384);            // x-projection
    k_gru_seq<<<NB / 16, 256, 0, stream>>>(xproj, whh0_16, bhh0, o1_32, o1_16);

    // ---- GRU layer 1 ----
    k_gemm_wmma<float, 128, 1><<<N_GRAPHS / 16, 256, 0, stream>>>(
        o1_16, wih1_16, bih1, nullptr, 0, xproj, 384);
    k_gru_seq<<<NB / 16, 256, 0, stream>>>(xproj, whh1_16, bhh1, o2_32, o2_16);

    // ---- decoder: 4096 predictions + 256 final ----
    k_decoder<<<(N_GRAPHS + NB + 7) / 8, 256, 0, stream>>>(o2_32, w_dec, b_dec, out);
}
